// LlamaSmartKVAttention_50019189129805
// MI455X (gfx1250) — matrix operand order repaired and checked by reading, not compiled, not run on hardware
//
#include <hip/hip_runtime.h>
#include <hip/hip_bf16.h>
#include <stdint.h>

// ---------------- problem constants ----------------
#define H    32
#define HKV  8
#define G    (H / HKV)      // 4
#define D    128
#define HID  4096
#define S    32768

#define NSPLIT   256
#define SPLITLEN (S / NSPLIT)   // 128

#define NQ (H * D)     // 4096
#define NK (HKV * D)   // 1024

typedef __attribute__((ext_vector_type(16))) _Float16 v16h;
typedef __attribute__((ext_vector_type(8)))  float    v8f;
typedef __attribute__((ext_vector_type(4)))  unsigned int v4u;
typedef __attribute__((ext_vector_type(4)))  int      v4i;
typedef __attribute__((ext_vector_type(8)))  int      v8i;

// Tensor Data Mover path, guarded so the file also compiles on toolchains
// without the builtin (and on the host pass). This toolchain exposes the
// 6-arg form: (uint32x4 g0, int32x8 g1, int32x4, int32x4, int32x8, i32 cpol).
#if defined(__has_builtin)
#if __has_builtin(__builtin_amdgcn_tensor_load_to_lds) && \
    __has_builtin(__builtin_amdgcn_s_wait_tensorcnt)
#define USE_TDM 1
#endif
#endif
#ifndef USE_TDM
#define USE_TDM 0
#endif

// A-matrix (16-bit, 16x32) element->K mapping per ISA 7.12.2:
// VGPR0..3 hold K = hi*8 + {0..7}; VGPR4..7 hold K = 16 + hi*8 + {0..7}
__device__ __forceinline__ int a_k_of(int e, int hi) {
    return (e < 8) ? (hi * 8 + e) : (16 + hi * 8 + (e - 8));
}

#if USE_TDM
// ---- D# group 0 (ISA 8.3): count=1 | lds_addr<<32 ; global_addr | type=2 ----
__device__ __forceinline__ v4u tdm_group0(unsigned lds_addr, unsigned long long gaddr) {
    union { unsigned long long q[2]; v4u v; } u;
    u.q[0] = 1ull | ((unsigned long long)lds_addr << 32);
    u.q[1] = (gaddr & ((1ull << 57) - 1ull)) | (2ull << 62);
    return u.v;
}
// ---- D# group 1 (ISA 8.4): 2D tile, data_size=4B, no padding/iterate ----
__device__ __forceinline__ v8i tdm_group1(unsigned tdim0, unsigned tdim1,
                                          unsigned tile0, unsigned tile1,
                                          unsigned long long stride0) {
    union { unsigned long long q[4]; v8i v; } u;
    u.q[0] = ((unsigned long long)2u << 16)                          // data_size = 4B
           | (((unsigned long long)tdim0 & 0xFFFFull) << 48);        // tensor_dim0[15:0]
    u.q[1] = (((unsigned long long)tdim0 >> 16) & 0xFFFFull)         // tensor_dim0[31:16]
           | ((unsigned long long)tdim1 << 16)                       // tensor_dim1[31:0]
           | (((unsigned long long)tile0 & 0xFFFFull) << 48);        // tile_dim0
    u.q[2] = ((unsigned long long)tile1 & 0xFFFFull)                 // tile_dim1
           | ((stride0 & 0xFFFFFFFFull) << 32);                      // dim0_stride[31:0]
    u.q[3] = (stride0 >> 32) & 0xFFFFull;                            // dim0_stride[47:32]
    return u.v;
}
__device__ __forceinline__ void tdm_issue(const int* gsrc, const int* lds_dst, v8i g1) {
    v4u g0 = tdm_group0((unsigned)(size_t)lds_dst, (unsigned long long)(size_t)gsrc);
    const v4i z4 = (v4i){0, 0, 0, 0};
    const v8i z8 = (v8i){0, 0, 0, 0, 0, 0, 0, 0};
    __builtin_amdgcn_tensor_load_to_lds(g0, g1, z4, z4, z8, 0);
}
#endif

// synchronous staging fallback: rows stride HKV*D elements, 128 ints per row
__device__ __forceinline__ void stage_rows(int* dst, const int* src0, int rows, int lane) {
    for (int p = 0; p < rows; ++p)
        ((int4*)dst)[p * 32 + lane] = ((const int4*)(src0 + (size_t)p * (HKV * D)))[lane];
}

// ---------------- kernel 1: QKV GEMV + RoPE ----------------
__global__ __launch_bounds__(256) void qkv_rope_kernel(
    const float* __restrict__ hidden,
    const float* __restrict__ Wq, const float* __restrict__ Wk,
    const float* __restrict__ Wv,
    const float* __restrict__ cosv, const float* __restrict__ sinv,
    _Float16* __restrict__ qf, float* __restrict__ kcur, float* __restrict__ vcur)
{
    __shared__ float hs[HID];      // 16 KB
    __shared__ float tb[256];
    const int tid = threadIdx.x;
    for (int i = tid; i < HID; i += 256) hs[i] = hidden[i];
    __syncthreads();

    const int j = blockIdx.x * 256 + tid;   // kind boundaries are multiples of 256
    const float* W; int col, ncols, kind;
    if (j < NQ)           { W = Wq; col = j;           ncols = NQ; kind = 0; }
    else if (j < NQ + NK) { W = Wk; col = j - NQ;      ncols = NK; kind = 1; }
    else                  { W = Wv; col = j - NQ - NK; ncols = NK; kind = 2; }

    float acc = 0.f;
#pragma unroll 4
    for (int i = 0; i < HID; ++i)
        acc += hs[i] * W[(size_t)i * ncols + col];

    tb[tid] = acc;
    __syncthreads();

    const int d = col & (D - 1);
    float res = acc;
    if (kind != 2) {  // RoPE
        const int pj = (d < D / 2) ? (tid + D / 2) : (tid - D / 2);
        const float partner = tb[pj];
        const float rot = (d < D / 2) ? -partner : partner;
        res = acc * cosv[d] + rot * sinv[d];
    }
    if (kind == 0)      qf[col]   = (_Float16)(res * 0.08838834764831845f); // 1/sqrt(D)
    else if (kind == 1) kcur[col] = res;
    else                vcur[col] = res;
}

// ---------------- kernel 2: flash-decode split over S ----------------
// grid (NSPLIT, HKV), 1 wave per block; TDM double-buffered K/V tiles.
__global__ __launch_bounds__(32) void attn_split_kernel(
    const _Float16* __restrict__ qf,
    const int*   __restrict__ k_qx, const float* __restrict__ k_scale,
    const int*   __restrict__ v_qx, const float* __restrict__ v_scale,
    float* __restrict__ part_m, float* __restrict__ part_l,
    float* __restrict__ part_acc)
{
    const int split = blockIdx.x;
    const int kh    = blockIdx.y;
    const int lane  = threadIdx.x;
    const int hi    = lane >> 4;
    const int ln    = lane & 15;
    const int s0    = split * SPLITLEN;

    __shared__ int      ktile[2][16 * D];   // 16 KB double-buffered K tiles
    __shared__ int      vtile[2][32 * D];   // 32 KB double-buffered V chunks
    __shared__ float    sc[G][SPLITLEN];    //  2 KB scores
    __shared__ _Float16 pb[G][SPLITLEN];    //  1 KB probs

#if USE_TDM
    const v8i g1k = tdm_group1(D, 1u << 20, D, 16, (unsigned long long)(HKV * D));
    const v8i g1v = tdm_group1(D, 1u << 20, D, 32, (unsigned long long)(HKV * D));
#endif

    // ---- kick off first K tile (overlaps with building q fragments) ----
    {
        const int* src = k_qx + ((size_t)s0 * HKV + kh) * D;
#if USE_TDM
        tdm_issue(src, &ktile[0][0], g1k);
#else
        stage_rows(&ktile[0][0], src, 16, lane);
#endif
    }

    // ---- q A-fragments: M = 4 heads (padded to 16), K = D = 4 chunks of 32 ----
    v16h afrag[4];
#pragma unroll
    for (int c = 0; c < 4; ++c) {
#pragma unroll
        for (int e = 0; e < 16; ++e) {
            const int k = c * 32 + a_k_of(e, hi);
            _Float16 val = (_Float16)0.f;
            if (ln < G) val = qf[(kh * G + ln) * D + k];
            afrag[c][e] = val;
        }
    }

    // ---- QK^T over 8 tiles of 16 positions, pipelined ----
    const int NTK = SPLITLEN / 16;   // 8
    for (int tile = 0; tile < NTK; ++tile) {
        const int cur = tile & 1;
        if (tile + 1 < NTK) {
            __syncthreads();   // drain reads of the buffer being overwritten
            const int* src = k_qx + ((size_t)(s0 + (tile + 1) * 16) * HKV + kh) * D;
#if USE_TDM
            tdm_issue(src, &ktile[(tile + 1) & 1][0], g1k);
#else
            __builtin_prefetch(src, 0, 0);
            stage_rows(&ktile[(tile + 1) & 1][0], src, 16, lane);
#endif
        }
#if USE_TDM
        if (tile + 1 < NTK) __builtin_amdgcn_s_wait_tensorcnt(1);
        else                __builtin_amdgcn_s_wait_tensorcnt(0);
#else
        __syncthreads();
#endif
        const float scl = k_scale[(size_t)(s0 + tile * 16 + ln) * HKV + kh]; // n = position
        v8f cacc = {};
#pragma unroll
        for (int c = 0; c < 4; ++c) {
            v16h bfrag;   // B: n = position (ln), K = dim; same K order as A (a_k_of)
            const int kbase = c * 32 + hi * 16;
#pragma unroll
            for (int e = 0; e < 16; ++e)
                bfrag[e] = (_Float16)((float)ktile[cur][ln * D + c * 32 + a_k_of(e, hi)] * scl);
            cacc = __builtin_amdgcn_wmma_f32_16x16x32_f16(
                       false, afrag[c], false, bfrag, (short)0, cacc, false, false);
        }
        if (hi == 0) {
#pragma unroll
            for (int r = 0; r < G; ++r)
                sc[r][tile * 16 + ln] = cacc[r];
        }
    }
    __syncthreads();   // sc visible to all lanes

    // ---- kick off first V chunk; it streams in during softmax ----
    {
        const int* src = v_qx + ((size_t)s0 * HKV + kh) * D;
#if USE_TDM
        tdm_issue(src, &vtile[0][0], g1v);
#else
        stage_rows(&vtile[0][0], src, 32, lane);
#endif
    }

    // ---- local softmax ----
    const int g  = lane >> 3;
    const int sl = lane & 7;
    float mx = -3.0e38f;
    for (int t = sl; t < SPLITLEN; t += 8) mx = fmaxf(mx, sc[g][t]);
#pragma unroll
    for (int off = 1; off < 8; off <<= 1) mx = fmaxf(mx, __shfl_xor(mx, off, 32));
    float sum = 0.f;
    for (int t = sl; t < SPLITLEN; t += 8) {
        const float e = __expf(sc[g][t] - mx);
        pb[g][t] = (_Float16)e;
        sum += e;
    }
#pragma unroll
    for (int off = 1; off < 8; off <<= 1) sum += __shfl_xor(sum, off, 32);
    if (sl == 0) {
        part_m[(kh * G + g) * NSPLIT + split] = mx;
        part_l[(kh * G + g) * NSPLIT + split] = sum;
    }
    __syncthreads();   // pb visible

    // ---- P @ V : K = 128 positions (4 chunks of 32), N = 128 dims (8 tiles), pipelined ----
    const int NTV = SPLITLEN / 32;   // 4
    v8f oacc[8];
#pragma unroll
    for (int t = 0; t < 8; ++t) oacc[t] = (v8f){};
    for (int kc = 0; kc < NTV; ++kc) {
        const int cur = kc & 1;
        if (kc + 1 < NTV) {
            __syncthreads();
            const int* src = v_qx + ((size_t)(s0 + (kc + 1) * 32) * HKV + kh) * D;
#if USE_TDM
            tdm_issue(src, &vtile[(kc + 1) & 1][0], g1v);
#else
            __builtin_prefetch(src, 0, 0);
            stage_rows(&vtile[(kc + 1) & 1][0], src, 32, lane);
#endif
        }
#if USE_TDM
        if (kc + 1 < NTV) __builtin_amdgcn_s_wait_tensorcnt(1);
        else              __builtin_amdgcn_s_wait_tensorcnt(0);
#else
        __syncthreads();
#endif
        // per-lane-half V scales via shuffle broadcast (no LDS round-trip)
        const float svl = v_scale[(size_t)(s0 + kc * 32 + lane) * HKV + kh];
        float svv[16];
#pragma unroll
        for (int e = 0; e < 16; ++e) svv[e] = __shfl(svl, a_k_of(e, hi), 32);

        v16h pa;   // A from probs: M = heads, K = 32 positions of this chunk
#pragma unroll
        for (int e = 0; e < 16; ++e) {
            const int k = a_k_of(e, hi);
            _Float16 val = (_Float16)0.f;
            if (ln < G) val = pb[ln][kc * 32 + k];
            pa[e] = val;
        }
#pragma unroll
        for (int t = 0; t < 8; ++t) {
            v16h vb;   // B: n = dim (t*16+ln), K = position in chunk
#pragma unroll
            for (int e = 0; e < 16; ++e) {
                const int kpos = a_k_of(e, hi);
                vb[e] = (_Float16)((float)vtile[cur][kpos * D + t * 16 + ln] * svv[e]);
            }
            oacc[t] = __builtin_amdgcn_wmma_f32_16x16x32_f16(
                          false, pa, false, vb, (short)0, oacc[t], false, false);
        }
    }

    if (hi == 0) {
#pragma unroll
        for (int t = 0; t < 8; ++t)
#pragma unroll
            for (int r = 0; r < G; ++r)
                part_acc[(((size_t)(kh * G + r)) * NSPLIT + split) * D + t * 16 + ln] = oacc[t][r];
    }
}

// ---------------- kernel 3: combine splits + current-token term ----------------
__global__ __launch_bounds__(D) void attn_reduce_kernel(
    const _Float16* __restrict__ qf,
    const float* __restrict__ kcur, const float* __restrict__ vcur,
    const float* __restrict__ part_m, const float* __restrict__ part_l,
    const float* __restrict__ part_acc,
    float* __restrict__ attn)
{
    const int h  = blockIdx.x;
    const int d  = threadIdx.x;
    const int kh = h / G;

    __shared__ float red[D];
    red[d] = (float)qf[h * D + d] * kcur[kh * D + d];   // qf has 1/sqrt(D) baked in
    __syncthreads();
    for (int off = D / 2; off > 0; off >>= 1) {
        if (d < off) red[d] += red[d + off];
        __syncthreads();
    }
    const float scur = red[0];

    float M = scur;
    for (int i = 0; i < NSPLIT; ++i) M = fmaxf(M, part_m[h * NSPLIT + i]);

    float l   = __expf(scur - M);
    float acc = l * vcur[kh * D + d];
    for (int i = 0; i < NSPLIT; ++i) {
        const float w = __expf(part_m[h * NSPLIT + i] - M);
        l   += part_l[h * NSPLIT + i] * w;
        acc += part_acc[((size_t)h * NSPLIT + i) * D + d] * w;
    }
    attn[h * D + d] = acc / l;
}

// ---------------- kernel 4: O projection GEMV ----------------
__global__ __launch_bounds__(256) void oproj_kernel(
    const float* __restrict__ attn, const float* __restrict__ Wo,
    float* __restrict__ out)
{
    __shared__ float as[NQ];
    const int tid = threadIdx.x;
    for (int i = tid; i < NQ; i += 256) as[i] = attn[i];
    __syncthreads();

    const int j = blockIdx.x * 256 + tid;
    float acc = 0.f;
#pragma unroll 4
    for (int i = 0; i < NQ; ++i)
        acc += as[i] * Wo[(size_t)i * HID + j];
    out[j] = acc;
}

// ---------------- host launcher ----------------
extern "C" void kernel_launch(void* const* d_in, const int* in_sizes, int n_in,
                              void* d_out, int out_size, void* d_ws, size_t ws_size,
                              hipStream_t stream)
{
    const float* hidden  = (const float*)d_in[0];
    const int*   k_qx    = (const int*)  d_in[1];
    const float* k_scale = (const float*)d_in[2];
    const int*   v_qx    = (const int*)  d_in[3];
    const float* v_scale = (const float*)d_in[4];
    const float* cosv    = (const float*)d_in[5];
    const float* sinv    = (const float*)d_in[6];
    const float* Wq      = (const float*)d_in[7];
    const float* Wk      = (const float*)d_in[8];
    const float* Wv      = (const float*)d_in[9];
    const float* Wo      = (const float*)d_in[10];
    float* out = (float*)d_out;

    char* ws = (char*)d_ws;
    _Float16* qf     = (_Float16*)(ws + 0);                 //  8 KB
    float*    kcur   = (float*)(ws + 8192);                 //  4 KB
    float*    vcur   = (float*)(ws + 12288);                //  4 KB
    float*    part_m = (float*)(ws + 16384);                // 32 KB
    float*    part_l = (float*)(ws + 49152);                // 32 KB
    float*    attn   = (float*)(ws + 81920);                // 16 KB
    float*    pacc   = (float*)(ws + 98304);                //  4 MB

    qkv_rope_kernel<<<(NQ + 2 * NK) / 256, 256, 0, stream>>>(
        hidden, Wq, Wk, Wv, cosv, sinv, qf, kcur, vcur);

    attn_split_kernel<<<dim3(NSPLIT, HKV, 1), 32, 0, stream>>>(
        qf, k_qx, k_scale, v_qx, v_scale, part_m, part_l, pacc);

    attn_reduce_kernel<<<H, D, 0, stream>>>(
        qf, kcur, vcur, part_m, part_l, pacc, attn);

    oproj_kernel<<<HID / 256, 256, 0, stream>>>(attn, Wo, out);

    (void)in_sizes; (void)n_in; (void)out_size; (void)ws_size;
}
